// swin_large_patch4_window12_384_in22k_58085137711741
// MI455X (gfx1250) — matrix operand order repaired and checked by reading, not compiled, not run on hardware
//
#include <hip/hip_runtime.h>

typedef __bf16 bh;
typedef __attribute__((ext_vector_type(4)))  __bf16 v4bh;
typedef __attribute__((ext_vector_type(8)))  __bf16 v8bh;
typedef __attribute__((ext_vector_type(16))) __bf16 v16bh;
typedef __attribute__((ext_vector_type(8)))  float  v8f;

static __device__ __forceinline__ v16bh cat8(v8bh lo, v8bh hi) {
  return __builtin_shufflevector(lo, hi, 0,1,2,3,4,5,6,7,8,9,10,11,12,13,14,15);
}
static __device__ __forceinline__ v8f wmma_bf16(v16bh a, v16bh b, v8f c) {
  return __builtin_amdgcn_wmma_f32_16x16x32_bf16(false, a, false, b, (short)0, c, false, false);
}
// A-fragment: lane holds row (lane&15); K halves at +0..7 and +16..23 (lane<16) or shifted by 8 (lane>=16).
static __device__ __forceinline__ v16bh ldfragA(const bh* p) {
  v8bh lo = *(const v8bh*)p;
  v8bh hi = *(const v8bh*)(p + 16);
  return cat8(lo, hi);
}
// B-fragment: lane holds column (lane&15); K halves at +0..7 and +16..23 (lane<16) or shifted by 8 (lane>=16).
static __device__ __forceinline__ v16bh ldfragB(const bh* p) {
  v8bh lo = *(const v8bh*)p;
  v8bh hi = *(const v8bh*)(p + 16);
  return cat8(lo, hi);
}

#define WIN 12
#define SHIFT 6
#define HEADS 24
#define NTOK 144            // win*win
#define NTOKP 160           // padded K for attn@v
#define CDIM 768
#define DHEAD 32
#define NROWS 36864         // 256 windows * 144 tokens
#define QKVN 2304
#define LDSTR 40            // LDS row stride (halves) for 32-wide K tiles

// ---------------------------------------------------------------------------
// Prep: transpose f32 [K][N] -> bf16 [N][K]
// ---------------------------------------------------------------------------
__global__ __launch_bounds__(256) void transpose_conv_kernel(const float* __restrict__ src,
                                                             bh* __restrict__ dst,
                                                             int K, int N) {
  int idx = blockIdx.x * 256 + threadIdx.x;
  if (idx >= K * N) return;
  int n = idx / K;
  int k = idx - n * K;
  dst[idx] = (bh)src[(size_t)k * N + n];
}

// ---------------------------------------------------------------------------
// QKV GEMM: A = shifted-window-gathered x (f32 -> bf16), B = qkv_wT bf16 [2304][768]
// C tile 128x128, K-step 32.  Epilogue: +bias, q*=scale, scatter to q/k/vT bf16.
// ---------------------------------------------------------------------------
__global__ __launch_bounds__(256) void gemm_qkv_kernel(const float* __restrict__ x,
                                                       const bh* __restrict__ wT,
                                                       const float* __restrict__ qkv_b,
                                                       bh* __restrict__ qbuf,
                                                       bh* __restrict__ kbuf,
                                                       bh* __restrict__ vtbuf) {
  __shared__ bh As[128 * LDSTR];
  __shared__ bh Bs[128 * LDSTR];
  const int t = threadIdx.x;
  const int m0 = blockIdx.y * 128;
  const int n0 = blockIdx.x * 128;
  const int wid = t >> 5, lane = t & 31;
  const int lrow = lane & 15, lhi = lane >> 4;
  const int wr = wid & 3;       // row group (4 x 32 rows)
  const int wc = wid >> 2;      // col group (2 x 64 cols)

  // Precompute gathered row bases for the 4 A rows this thread loads.
  int gb[4];
  const int q8 = t & 7;
#pragma unroll
  for (int p = 0; p < 4; ++p) {
    int r = (t >> 3) + 32 * p;
    int m = m0 + r;
    int bw = m / 144;
    int n = m - bw * 144;
    int b = bw >> 2, w = bw & 3;
    int wh = w >> 1, ww = w & 1;
    int i = n / 12, j = n - (n / 12) * 12;
    int hs = wh * 12 + i + SHIFT; if (hs >= 24) hs -= 24;
    int ws = ww * 12 + j + SHIFT; if (ws >= 24) ws -= 24;
    gb[p] = ((b * 24 + hs) * 24 + ws) * CDIM + q8 * 4;
  }
  const int rb = t >> 1, hb = t & 1;   // B loader: 128 rows x 32 halves

  v8f acc[2][4];
#pragma unroll
  for (int ai = 0; ai < 2; ++ai)
#pragma unroll
    for (int bi = 0; bi < 4; ++bi) acc[ai][bi] = (v8f){0.f,0.f,0.f,0.f,0.f,0.f,0.f,0.f};

  for (int kt = 0; kt < CDIM / 32; ++kt) {
    __syncthreads();
    // A tile: gather 128x32 f32, convert to bf16
#pragma unroll
    for (int p = 0; p < 4; ++p) {
      int r = (t >> 3) + 32 * p;
      float4 f = *(const float4*)(x + gb[p] + kt * 32);
      v4bh h = {(bh)f.x, (bh)f.y, (bh)f.z, (bh)f.w};
      *(v4bh*)(As + r * LDSTR + q8 * 4) = h;
    }
    // B tile: 128x32 bf16 from wT
    {
      const bh* src = wT + (size_t)(n0 + rb) * CDIM + kt * 32 + hb * 16;
      *(v8bh*)(Bs + rb * LDSTR + hb * 16)     = *(const v8bh*)src;
      *(v8bh*)(Bs + rb * LDSTR + hb * 16 + 8) = *(const v8bh*)(src + 8);
    }
    __syncthreads();

    v16bh bfr[4];
#pragma unroll
    for (int bi = 0; bi < 4; ++bi)
      bfr[bi] = ldfragB(Bs + (wc * 64 + bi * 16 + lrow) * LDSTR + lhi * 8);
#pragma unroll
    for (int ai = 0; ai < 2; ++ai) {
      v16bh afr = ldfragA(As + (wr * 32 + ai * 16 + lrow) * LDSTR + lhi * 8);
#pragma unroll
      for (int bi = 0; bi < 4; ++bi)
        acc[ai][bi] = wmma_bf16(afr, bfr[bi], acc[ai][bi]);
    }
  }

  // Epilogue: bias, scale q, scatter bf16.
  const float scale = 0.17677669529663687f;  // 32^-0.5
#pragma unroll
  for (int ai = 0; ai < 2; ++ai) {
#pragma unroll
    for (int bi = 0; bi < 4; ++bi) {
      int gn = n0 + wc * 64 + bi * 16 + lrow;
      float bias = qkv_b[gn];
      int which = gn / CDIM;
      int rem = gn - which * CDIM;
      int hd = rem >> 5, ch = rem & 31;
      int gm_base = m0 + wr * 32 + ai * 16 + 8 * lhi;
#pragma unroll
      for (int r = 0; r < 8; ++r) {
        int gm = gm_base + r;
        float val = acc[ai][bi][r] + bias;
        if (which == 0) val *= scale;
        int bw = gm / 144;
        int n = gm - bw * 144;
        bh hv = (bh)val;
        if (which == 0)
          qbuf[((size_t)(bw * HEADS + hd) * NTOK + n) * DHEAD + ch] = hv;
        else if (which == 1)
          kbuf[((size_t)(bw * HEADS + hd) * NTOK + n) * DHEAD + ch] = hv;
        else
          vtbuf[((size_t)(bw * HEADS + hd) * DHEAD + ch) * NTOKP + n] = hv;
      }
    }
  }
}

// ---------------------------------------------------------------------------
// Attention: one block per (window, head), 9 waves; each wave = 16 query rows.
// q@k^T (9 WMMA) -> +rel-pos bias +shift mask -> softmax -> attn@v (10 WMMA).
// ---------------------------------------------------------------------------
__global__ __launch_bounds__(288) void attn_kernel(const bh* __restrict__ qb,
                                                   const bh* __restrict__ kb,
                                                   const bh* __restrict__ vtb,
                                                   const float* __restrict__ rpb_table,
                                                   bh* __restrict__ attnout) {
  __shared__ bh At[9 * 16 * NTOKP];   // per-wave bf16 attn staging (K padded to 160)
  __shared__ float rpbs[529];
  const int blk = blockIdx.x;
  const int h = blk % HEADS;
  const int bw = blk / HEADS;
  const int w = bw & 3;
  const int wh = w >> 1, ww = w & 1;
  const int t = threadIdx.x;
  const int wid = t >> 5, lane = t & 31;
  const int lrow = lane & 15, lhi = lane >> 4;
  const int m0 = wid * 16;

  for (int idx = t; idx < 529; idx += 288) rpbs[idx] = rpb_table[idx * HEADS + h];

  bh* myAt = At + wid * 16 * NTOKP;
  { v8bh z = {}; *(v8bh*)(myAt + lrow * NTOKP + NTOK + lhi * 8) = z; }  // zero pad K=144..159

  const bh* qbase  = qb  + (size_t)(bw * HEADS + h) * NTOK * DHEAD;
  const bh* kbase  = kb  + (size_t)(bw * HEADS + h) * NTOK * DHEAD;
  const bh* vtbase = vtb + (size_t)(bw * HEADS + h) * DHEAD * NTOKP;

  // q fragment: this wave's 16 rows, K = 0..31 (d = 32, one K-step)
  const bh* qp = qbase + (m0 + lrow) * DHEAD + lhi * 8;
  v16bh afrag = ldfragA(qp);

  v8f acc[9];
#pragma unroll
  for (int nt = 0; nt < 9; ++nt) acc[nt] = (v8f){0.f,0.f,0.f,0.f,0.f,0.f,0.f,0.f};
#pragma unroll
  for (int nt = 0; nt < 9; ++nt) {
    const bh* kp = kbase + (nt * 16 + lrow) * DHEAD + lhi * 8;
    acc[nt] = wmma_bf16(afrag, ldfragB(kp), acc[nt]);
  }

  __syncthreads();  // rpbs ready

  // query-token coordinates / mask region ids for the 8 rows this lane owns
  int yiA[8], xiA[8], idiA[8];
#pragma unroll
  for (int r = 0; r < 8; ++r) {
    int i = m0 + r + 8 * lhi;
    int yi = i / 12, xi = i - yi * 12;
    yiA[r] = yi; xiA[r] = xi;
    int ri = wh ? ((yi < SHIFT) ? 1 : 2) : 0;
    int ci = ww ? ((xi < SHIFT) ? 1 : 2) : 0;
    idiA[r] = ri * 3 + ci;
  }

  // bias + mask
#pragma unroll
  for (int nt = 0; nt < 9; ++nt) {
    int j = nt * 16 + lrow;
    int yj = j / 12, xj = j - (j / 12) * 12;
    int rj = wh ? ((yj < SHIFT) ? 1 : 2) : 0;
    int cj = ww ? ((xj < SHIFT) ? 1 : 2) : 0;
    int idj = rj * 3 + cj;
#pragma unroll
    for (int r = 0; r < 8; ++r) {
      int rel = (yiA[r] - yj + 11) * 23 + (xiA[r] - xj + 11);
      float bm = rpbs[rel] + ((idiA[r] != idj) ? -100.0f : 0.0f);
      acc[nt][r] += bm;
    }
  }

  // softmax across 144 keys: per (row, half-wave) butterfly over 16 lanes
  float mx[8];
#pragma unroll
  for (int r = 0; r < 8; ++r) {
    float m = acc[0][r];
#pragma unroll
    for (int nt = 1; nt < 9; ++nt) m = fmaxf(m, acc[nt][r]);
    mx[r] = m;
  }
#pragma unroll
  for (int d = 8; d >= 1; d >>= 1) {
#pragma unroll
    for (int r = 0; r < 8; ++r) mx[r] = fmaxf(mx[r], __shfl_xor(mx[r], d, 32));
  }
  float sm[8];
#pragma unroll
  for (int r = 0; r < 8; ++r) sm[r] = 0.0f;
#pragma unroll
  for (int nt = 0; nt < 9; ++nt) {
#pragma unroll
    for (int r = 0; r < 8; ++r) {
      float e = __expf(acc[nt][r] - mx[r]);
      acc[nt][r] = e;
      sm[r] += e;
    }
  }
#pragma unroll
  for (int d = 8; d >= 1; d >>= 1) {
#pragma unroll
    for (int r = 0; r < 8; ++r) sm[r] += __shfl_xor(sm[r], d, 32);
  }
  float inv[8];
#pragma unroll
  for (int r = 0; r < 8; ++r) inv[r] = 1.0f / sm[r];

  // stage normalized probabilities as bf16 in A-layout-friendly LDS rows
#pragma unroll
  for (int nt = 0; nt < 9; ++nt) {
#pragma unroll
    for (int r = 0; r < 8; ++r)
      myAt[(r + 8 * lhi) * NTOKP + nt * 16 + lrow] = (bh)(acc[nt][r] * inv[r]);
  }
#if __has_builtin(__builtin_amdgcn_s_wait_dscnt)
  __builtin_amdgcn_s_wait_dscnt(0);
#else
  __syncthreads();
#endif

  // attn @ v : K = 160 (padded), N = 32 channels
  v8f oacc[2];
  oacc[0] = (v8f){0.f,0.f,0.f,0.f,0.f,0.f,0.f,0.f};
  oacc[1] = (v8f){0.f,0.f,0.f,0.f,0.f,0.f,0.f,0.f};
#pragma unroll
  for (int kc = 0; kc < 5; ++kc) {
    v16bh af = ldfragA(myAt + lrow * NTOKP + kc * 32 + lhi * 8);
#pragma unroll
    for (int nc = 0; nc < 2; ++nc) {
      const bh* vp = vtbase + (nc * 16 + lrow) * NTOKP + kc * 32 + lhi * 8;
      oacc[nc] = wmma_bf16(af, ldfragB(vp), oacc[nc]);
    }
  }

  // write head-concatenated output rows (bf16) for the projection GEMM
  size_t rowbase = (size_t)bw * NTOK;
#pragma unroll
  for (int nc = 0; nc < 2; ++nc) {
#pragma unroll
    for (int r = 0; r < 8; ++r) {
      int m = m0 + r + 8 * lhi;
      int c = h * DHEAD + nc * 16 + lrow;
      attnout[(rowbase + m) * CDIM + c] = (bh)oacc[nc][r];
    }
  }
}

// ---------------------------------------------------------------------------
// Proj GEMM: A = attnout bf16 [36864][768], B = proj_wT bf16 [768][768] -> f32
// ---------------------------------------------------------------------------
__global__ __launch_bounds__(256) void gemm_proj_kernel(const bh* __restrict__ a,
                                                        const bh* __restrict__ wT,
                                                        float* __restrict__ out) {
  __shared__ bh As[128 * LDSTR];
  __shared__ bh Bs[128 * LDSTR];
  const int t = threadIdx.x;
  const int m0 = blockIdx.y * 128;
  const int n0 = blockIdx.x * 128;
  const int wid = t >> 5, lane = t & 31;
  const int lrow = lane & 15, lhi = lane >> 4;
  const int wr = wid & 3, wc = wid >> 2;
  const int rb = t >> 1, hb = t & 1;

  v8f acc[2][4];
#pragma unroll
  for (int ai = 0; ai < 2; ++ai)
#pragma unroll
    for (int bi = 0; bi < 4; ++bi) acc[ai][bi] = (v8f){0.f,0.f,0.f,0.f,0.f,0.f,0.f,0.f};

  for (int kt = 0; kt < CDIM / 32; ++kt) {
    __syncthreads();
    {
      const bh* src = a + (size_t)(m0 + rb) * CDIM + kt * 32 + hb * 16;
      *(v8bh*)(As + rb * LDSTR + hb * 16)     = *(const v8bh*)src;
      *(v8bh*)(As + rb * LDSTR + hb * 16 + 8) = *(const v8bh*)(src + 8);
    }
    {
      const bh* src = wT + (size_t)(n0 + rb) * CDIM + kt * 32 + hb * 16;
      *(v8bh*)(Bs + rb * LDSTR + hb * 16)     = *(const v8bh*)src;
      *(v8bh*)(Bs + rb * LDSTR + hb * 16 + 8) = *(const v8bh*)(src + 8);
    }
    __syncthreads();

    v16bh bfr[4];
#pragma unroll
    for (int bi = 0; bi < 4; ++bi)
      bfr[bi] = ldfragB(Bs + (wc * 64 + bi * 16 + lrow) * LDSTR + lhi * 8);
#pragma unroll
    for (int ai = 0; ai < 2; ++ai) {
      v16bh afr = ldfragA(As + (wr * 32 + ai * 16 + lrow) * LDSTR + lhi * 8);
#pragma unroll
      for (int bi = 0; bi < 4; ++bi)
        acc[ai][bi] = wmma_bf16(afr, bfr[bi], acc[ai][bi]);
    }
  }

#pragma unroll
  for (int ai = 0; ai < 2; ++ai) {
#pragma unroll
    for (int bi = 0; bi < 4; ++bi) {
      int gn = n0 + wc * 64 + bi * 16 + lrow;
      int gm_base = m0 + wr * 32 + ai * 16 + 8 * lhi;
#pragma unroll
      for (int r = 0; r < 8; ++r)
        out[(size_t)(gm_base + r) * CDIM + gn] = acc[ai][bi][r];
    }
  }
}

// ---------------------------------------------------------------------------
// Epilogue: +proj bias, L2 norm over C, window merge + reverse cyclic shift.
// One block per token row.
// ---------------------------------------------------------------------------
__global__ __launch_bounds__(256) void norm_kernel(const float* __restrict__ projout,
                                                   const float* __restrict__ proj_b,
                                                   float* __restrict__ out) {
  __shared__ float ps[8];
  const int m = blockIdx.x;
  const int t = threadIdx.x;
  int bw = m / 144;
  int n = m - bw * 144;
  int b = bw >> 2, w = bw & 3;
  int wh = w >> 1, ww = w & 1;
  int i = n / 12, j = n - (n / 12) * 12;
  int hs = wh * 12 + i + SHIFT; if (hs >= 24) hs -= 24;
  int ws = ww * 12 + j + SHIFT; if (ws >= 24) ws -= 24;
  size_t obase = ((size_t)(b * 24 + hs) * 24 + ws) * CDIM;
  const float* row = projout + (size_t)m * CDIM;

  float v0 = row[t]       + proj_b[t];
  float v1 = row[t + 256] + proj_b[t + 256];
  float v2 = row[t + 512] + proj_b[t + 512];
  float s = v0 * v0 + v1 * v1 + v2 * v2;
#pragma unroll
  for (int d = 16; d >= 1; d >>= 1) s += __shfl_xor(s, d, 32);
  if ((t & 31) == 0) ps[t >> 5] = s;
  __syncthreads();
  float tot = ps[0] + ps[1] + ps[2] + ps[3] + ps[4] + ps[5] + ps[6] + ps[7];
  float inv = rsqrtf(tot + 1e-6f);
  out[obase + t]       = v0 * inv;
  out[obase + t + 256] = v1 * inv;
  out[obase + t + 512] = v2 * inv;
}

// ---------------------------------------------------------------------------
extern "C" void kernel_launch(void* const* d_in, const int* in_sizes, int n_in,
                              void* d_out, int out_size, void* d_ws, size_t ws_size,
                              hipStream_t stream) {
  (void)in_sizes; (void)n_in; (void)out_size; (void)ws_size;
  const float* x         = (const float*)d_in[0];
  const float* qkv_w     = (const float*)d_in[1];
  const float* qkv_b     = (const float*)d_in[2];
  const float* proj_w    = (const float*)d_in[3];
  const float* proj_b    = (const float*)d_in[4];
  const float* rpb_table = (const float*)d_in[5];
  // d_in[6] (rel_index) recomputed inline.
  float* out = (float*)d_out;

  char* ws = (char*)d_ws;
  const size_t OFF_QKVWT  = 0;                                   // 2304*768*2   = 3,538,944
  const size_t OFF_PROJWT = OFF_QKVWT  + (size_t)QKVN * CDIM * 2; // 768*768*2
  const size_t OFF_Q      = OFF_PROJWT + (size_t)CDIM * CDIM * 2;
  const size_t SZ_QK      = (size_t)256 * HEADS * NTOK * DHEAD * 2;   // 56,623,104
  const size_t OFF_K      = OFF_Q + SZ_QK;
  const size_t OFF_VT     = OFF_K + SZ_QK;
  const size_t SZ_VT      = (size_t)256 * HEADS * DHEAD * NTOKP * 2;  // 62,914,560
  const size_t OFF_AO     = OFF_VT + SZ_VT;
  const size_t OFF_PROJO  = OFF_Q;  // alias: q+k dead once proj GEMM runs (exactly 113,246,208 B)

  bh*    qkv_wT  = (bh*)(ws + OFF_QKVWT);
  bh*    proj_wT = (bh*)(ws + OFF_PROJWT);
  bh*    qbuf    = (bh*)(ws + OFF_Q);
  bh*    kbuf    = (bh*)(ws + OFF_K);
  bh*    vtbuf   = (bh*)(ws + OFF_VT);
  bh*    attnout = (bh*)(ws + OFF_AO);
  float* projout = (float*)(ws + OFF_PROJO);

  // zero vT so K-pad tokens 144..159 contribute nothing
  hipMemsetAsync(vtbuf, 0, SZ_VT, stream);

  transpose_conv_kernel<<<(CDIM * QKVN + 255) / 256, 256, 0, stream>>>(qkv_w, qkv_wT, CDIM, QKVN);
  transpose_conv_kernel<<<(CDIM * CDIM + 255) / 256, 256, 0, stream>>>(proj_w, proj_wT, CDIM, CDIM);

  gemm_qkv_kernel<<<dim3(QKVN / 128, NROWS / 128), 256, 0, stream>>>(
      x, qkv_wT, qkv_b, qbuf, kbuf, vtbuf);

  attn_kernel<<<256 * HEADS, 288, 0, stream>>>(qbuf, kbuf, vtbuf, rpb_table, attnout);

  gemm_proj_kernel<<<dim3(CDIM / 128, NROWS / 128), 256, 0, stream>>>(
      attnout, proj_wT, projout);

  norm_kernel<<<NROWS, 256, 0, stream>>>(projout, proj_b, out);
}
